// NeuralCompGraphDecoder_13443247637286
// MI455X (gfx1250) — matrix operand order repaired and checked by reading, not compiled, not run on hardware
//
#include <hip/hip_runtime.h>
#include <hip/hip_bf16.h>

typedef __attribute__((ext_vector_type(16))) _Float16 v16h;
typedef __attribute__((ext_vector_type(8)))  _Float16 v8h;
typedef __attribute__((ext_vector_type(8)))  float    v8f;

namespace {

constexpr int kB        = 4096;
constexpr int kN        = 64;
constexpr int kSteps    = 2 * kN;                 // 128
constexpr int kTileRows = 16;
constexpr int kWaves    = 4;                      // waves per block, one tile each
constexpr int kTiles    = kB / kTileRows;         // 256
constexpr int kBlocks   = kTiles / kWaves;        // 64
constexpr int kEdgeScalars = 7 * 1024;            // 7168 scalars per batch row
constexpr int kTileF16  = kEdgeScalars * kTileRows; // 114688 f16 per tile

// d_out layout (float elements): logits(128,4096,4) | targets(128,4096) | u(64,4096) | v(64,4096)
constexpr int kOutTgt = kSteps * kB * 4;          // 2097152
constexpr int kOutU   = kOutTgt + kSteps * kB;    // 2621440
constexpr int kOutV   = kOutU + kN * kB;          // 2883584

// Weight region layout (f16 element offsets). Wt stored transposed: [Nout_pad][Kpad]
constexpr int WL1 = 0,     WL2 = 4096,  WL3 = 8192;     // calc_left  (48->64->64->16)
constexpr int WR1 = 9216,  WR2 = 13312, WR3 = 17408;    // calc_right
constexpr int WE1 = 18432, WE2 = 20480, WE3 = 24576;    // emb2logits (16->64->64->4)
constexpr int WG1 = 25600, WG2 = 27648, WG3 = 31744;    // logits2emb (4->64->64->16)
constexpr int kWTotal = 32768;                          // 64 KB f16
// Bias region (f32 element offsets), padded (hidden->64, final->16)
constexpr int BL1=0, BL2=64, BL3=128, BR1=144, BR2=208, BR3=272;
constexpr int BE1=288, BE2=352, BE3=416, BG1=432, BG2=496, BG3=560;
constexpr int kBTotal = 576;

__device__ __forceinline__ int edge_off(int beta) {
  // scalar offset (per batch row) of edge[beta]; level k node size = 64>>k, D=16
  int k = 31 - __builtin_clz((unsigned)beta);
  return k * 1024 + (beta - (1 << k)) * (64 >> k) * 16;
}

// A fragment (16x32 f16, M x K) from row-major LDS act buffer (pitch f16 elems).
// lanes 0-15: M=lane, K = k0+{0..7, 16..23}; lanes 16-31: M=lane-16, K = k0+{8..15, 24..31}
__device__ __forceinline__ v16h load_afrag(const _Float16* act, int pitch, int lane, int k0) {
  const int m  = lane & 15;
  const int hi = lane >> 4;
  const _Float16* r = act + m * pitch + k0 + hi * 8;
  v8h a0 = *(const v8h*)(r);
  v8h a1 = *(const v8h*)(r + 16);
  v16h o;
#pragma unroll
  for (int i = 0; i < 8; ++i) { o[i] = a0[i]; o[i + 8] = a1[i]; }
  return o;
}

// B fragment (32x16 f16, K x N) from transposed weights Wt[N][Kpad] in LDS.
// lanes 0-15: N=lane, K=k0+{0..7, 16..23} ; lanes 16-31: N=lane-16, K=k0+{8..15, 24..31}
__device__ __forceinline__ v16h load_bfrag(const _Float16* wt, int pitch, int lane, int nt, int k0) {
  const int n  = lane & 15;
  const int hi = lane >> 4;
  const _Float16* r = wt + (nt * 16 + n) * pitch + k0 + hi * 8;
  v8h b0 = *(const v8h*)(r);
  v8h b1 = *(const v8h*)(r + 16);
  v16h o;
#pragma unroll
  for (int i = 0; i < 8; ++i) { o[i] = b0[i]; o[i + 8] = b1[i]; }
  return o;
}

__device__ __forceinline__ float lse4f(float a, float b, float c, float d) {
  float m = fmaxf(fmaxf(a, b), fmaxf(c, d));
  return m + __logf(__expf(a - m) + __expf(b - m) + __expf(c - m) + __expf(d - m));
}
__device__ __forceinline__ void lsm4(float* x) {
  float m = fmaxf(fmaxf(x[0], x[1]), fmaxf(x[2], x[3]));
  float l = m + __logf(__expf(x[0]-m) + __expf(x[1]-m) + __expf(x[2]-m) + __expf(x[3]-m));
  x[0]-=l; x[1]-=l; x[2]-=l; x[3]-=l;
}

struct WPtrs { const float* p[24]; }; // (W,b) x 3 layers x {left,right,e2l,l2e}

} // namespace

// ---- prep: transpose + pad f32 weights into f16 Wt[N][Kpad] in ws; pad biases ----
__global__ void prep_weights(WPtrs wp, _Float16* __restrict__ wsW, float* __restrict__ wsB) {
  const int kin [12] = {48,64,64, 48,64,64, 16,64,64,  4,64,64};
  const int nout[12] = {64,64,16, 64,64,16, 64,64, 4, 64,64,16};
  const int kpad[12] = {64,64,64, 64,64,64, 32,64,64, 32,64,64};
  const int npad[12] = {64,64,16, 64,64,16, 64,64,16, 64,64,16};
  const int wdst[12] = {WL1,WL2,WL3, WR1,WR2,WR3, WE1,WE2,WE3, WG1,WG2,WG3};
  const int bdst[12] = {BL1,BL2,BL3, BR1,BR2,BR3, BE1,BE2,BE3, BG1,BG2,BG3};
  int tid = threadIdx.x + blockIdx.x * blockDim.x;
  int stride = blockDim.x * gridDim.x;
  for (int li = 0; li < 12; ++li) {
    const float* W = wp.p[2 * li];
    const float* b = wp.p[2 * li + 1];
    int sz = npad[li] * kpad[li];
    for (int t = tid; t < sz; t += stride) {
      int n = t / kpad[li], kk = t % kpad[li];
      float v = (kk < kin[li] && n < nout[li]) ? W[kk * nout[li] + n] : 0.0f;
      wsW[wdst[li] + t] = (_Float16)v;
    }
    for (int t = tid; t < npad[li]; t += stride)
      wsB[bdst[li] + t] = (t < nout[li]) ? b[t] : 0.0f;
  }
}

// ---- aux outputs: targets, u_out, v_out (pure reshuffles of inputs) ----
__global__ void fill_aux(const float* __restrict__ u, const float* __restrict__ v,
                         float* __restrict__ out) {
  int t = blockIdx.x * blockDim.x + threadIdx.x;
  if (t < kSteps * kB) {
    int step = t >> 12, bi = t & (kB - 1);
    int it = step >> 1; // i_t - 1
    out[kOutTgt + t] = u[bi * kN + it] * 2.0f + v[bi * kN + it];
  }
  if (t < kN * kB) {
    int i = t >> 12, bi = t & (kB - 1);
    out[kOutU + t] = u[bi * kN + i];
    out[kOutV + t] = v[bi * kN + i];
  }
}

// ---- persistent decoder: one wave per 16-row batch tile, full sequential schedule ----
__global__ void __launch_bounds__(kWaves * 32, 1)
polar_decoder(const int* __restrict__ zin, const float* __restrict__ ut,
              const float* __restrict__ vt, const float* __restrict__ embz,
              const float* __restrict__ noinfo,
              const _Float16* __restrict__ gW, const float* __restrict__ gB,
              _Float16* __restrict__ wsE, float* __restrict__ out) {
  __shared__ __align__(32) _Float16 sW[kWTotal];       // 64 KB weights
  __shared__ float sB[kBTotal];
  __shared__ float sEmb[48 * 16];
  __shared__ float sNI[16];
  __shared__ __align__(16) _Float16 sActA[kWaves][16 * 64];
  __shared__ __align__(16) _Float16 sActB[kWaves][16 * 64];
  __shared__ __align__(16) _Float16 sTmp [kWaves][256];
  __shared__ float sLL[kWaves][64];
  __shared__ float sRL[kWaves][64];
  __shared__ float sPF[kWaves][64];

  const int tid = threadIdx.x;
  const int wave = tid >> 5, lane = tid & 31;

  for (int i = tid; i < kWTotal; i += blockDim.x) sW[i] = gW[i];
  for (int i = tid; i < kBTotal; i += blockDim.x) sB[i] = gB[i];
  for (int i = tid; i < 48 * 16; i += blockDim.x) sEmb[i] = embz[i];
  if (tid < 16) sNI[tid] = noinfo[tid];
  __syncthreads();

  const int tile = blockIdx.x * kWaves + wave;
  const int b0 = tile * kTileRows;
  _Float16* tb = wsE + (size_t)tile * kTileF16;
  _Float16* actA = sActA[wave];
  _Float16* actB = sActB[wave];
  _Float16* tmpv = sTmp[wave];
  float* llb = sLL[wave];
  float* rlb = sRL[wave];
  float* pfb = sPF[wave];

  // ---- init edge tree: edge[1] = emb_z[z][:, bitrev]; all others = no_info ----
  for (int p = 0; p < 64; ++p) {
    int bp = __builtin_bitreverse32((unsigned)p) >> 26;
    if (lane < 16) {
      int zv = zin[(b0 + lane) * kN + bp];
#pragma unroll
      for (int d = 0; d < 16; ++d)
        tb[(p * 16 + d) * 16 + lane] = (_Float16)sEmb[zv * 16 + d];
    }
  }
  for (int j = 1024 * 16 + lane; j < kTileF16; j += 32)
    tb[j] = (_Float16)sNI[(j >> 4) & 15];

  // ---- fragment helpers (per-wave) ----
  auto stage_edge = [&](int eoff, _Float16* act, int pitch, int cb) {
    int d = lane & 15, hi = lane >> 4;
    v8h v = *(const v8h*)(tb + (size_t)eoff * 16 + d * 16 + hi * 8);
#pragma unroll
    for (int i = 0; i < 8; ++i) act[(hi * 8 + i) * pitch + cb + d] = v[i];
  };
  auto store_edge_c = [&](v8f c, int eoff) {
    int n = lane & 15, mb = (lane >> 4) * 8;
    v8h h;
#pragma unroll
    for (int r = 0; r < 8; ++r) h[r] = (_Float16)c[r];
    *(v8h*)(tb + (size_t)eoff * 16 + n * 16 + mb) = h;
  };
  auto store_c4 = [&](v8f c, float* buf) {
    int n = lane & 15, mb = (lane >> 4) * 8;
    if (n < 4) {
#pragma unroll
      for (int r = 0; r < 8; ++r) buf[(mb + r) * 4 + n] = c[r];
    }
  };
  auto layer_h = [&](const _Float16* ain, int kpad, _Float16* aout,
                     const _Float16* wt, const float* bias) {
    int n = lane & 15, mb = (lane >> 4) * 8;
    for (int nt = 0; nt < 4; ++nt) {
      float bv = bias[nt * 16 + n];
      v8f c;
#pragma unroll
      for (int r = 0; r < 8; ++r) c[r] = bv;
      for (int k0 = 0; k0 < kpad; k0 += 32) {
        v16h a = load_afrag(ain, kpad, lane, k0);
        v16h b = load_bfrag(wt, kpad, lane, nt, k0);
        c = __builtin_amdgcn_wmma_f32_16x16x32_f16(false, a, false, b, (short)0, c, false, false);
      }
#pragma unroll
      for (int r = 0; r < 8; ++r) {
        float x = c[r];
        x = x > 0.0f ? x : (__expf(x) - 1.0f);  // ELU
        aout[(mb + r) * 64 + nt * 16 + n] = (_Float16)x;
      }
    }
  };
  auto layer_f = [&](const _Float16* ain, const _Float16* wt, const float* bias) -> v8f {
    int n = lane & 15;
    float bv = bias[n];
    v8f c;
#pragma unroll
    for (int r = 0; r < 8; ++r) c[r] = bv;
    for (int k0 = 0; k0 < 64; k0 += 32) {
      v16h a = load_afrag(ain, 64, lane, k0);
      v16h b = load_bfrag(wt, 64, lane, 0, k0);
      c = __builtin_amdgcn_wmma_f32_16x16x32_f16(false, a, false, b, (short)0, c, false, false);
    }
    return c;
  };
  auto mlp3 = [&](int k1, int w1, int b1, int w2, int b2, int w3, int b3) -> v8f {
    layer_h(actA, k1, actB, sW + w1, sB + b1);
    layer_h(actB, 64, actA, sW + w2, sB + b2);
    return layer_f(actA, sW + w3, sB + b3);
  };

  // ---- tree ops ----
  auto calc_lr = [&](int beta, bool left) {
    int k = 31 - __builtin_clz((unsigned)beta);
    int l = 32 >> k;  // child size
    int poff = edge_off(beta);
    int meoff = edge_off(left ? 2 * beta : 2 * beta + 1);
    int otoff = edge_off(left ? 2 * beta + 1 : 2 * beta);
    int w1 = left ? WL1 : WR1, w2 = left ? WL2 : WR2, w3 = left ? WL3 : WR3;
    int b1 = left ? BL1 : BR1, b2 = left ? BL2 : BR2, b3 = left ? BL3 : BR3;
    for (int p = 0; p < l; ++p) {
      stage_edge(poff + p * 16, actA, 64, 0);
      stage_edge(poff + (l + p) * 16, actA, 64, 16);
      stage_edge(otoff + p * 16, actA, 64, 32);
      v8f c = mlp3(64, w1, b1, w2, b2, w3, b3);
      store_edge_c(c, meoff + p * 16);
    }
  };
  auto e2l_edge = [&](int eoff) -> v8f {
    stage_edge(eoff, actA, 32, 0);
    return mlp3(32, WE1, BE1, WE2, BE2, WE3, BE3);
  };
  auto l2e_from = [&](const float* buf) -> v8f {
    if (lane < 16) {
#pragma unroll
      for (int j = 0; j < 4; ++j) actA[lane * 32 + j] = (_Float16)buf[lane * 4 + j];
    }
    return mlp3(32, WG1, BG1, WG2, BG2, WG3, BG3);
  };
  auto calc_parent = [&](int beta) {
    int k = 31 - __builtin_clz((unsigned)beta);
    int l = 32 >> k;
    int boff = edge_off(beta), loff = edge_off(2 * beta), roff = edge_off(2 * beta + 1);
    for (int p = 0; p < l; ++p) {
      store_c4(e2l_edge(loff + p * 16), llb);
      store_c4(e2l_edge(roff + p * 16), rlb);
      if (lane < 16) {
        float a[4], b_[4];
#pragma unroll
        for (int j = 0; j < 4; ++j) { a[j] = llb[lane * 4 + j]; b_[j] = rlb[lane * 4 + j]; }
        lsm4(a); lsm4(b_);
        pfb[lane * 4 + 0] = lse4f(a[0]+b_[0], a[1]+b_[1], a[2]+b_[2], a[3]+b_[3]);
        pfb[lane * 4 + 1] = lse4f(a[1]+b_[0], a[0]+b_[1], a[3]+b_[2], a[2]+b_[3]);
        pfb[lane * 4 + 2] = lse4f(a[2]+b_[0], a[3]+b_[1], a[0]+b_[2], a[1]+b_[3]);
        pfb[lane * 4 + 3] = lse4f(a[3]+b_[0], a[2]+b_[1], a[1]+b_[2], a[0]+b_[3]);
#pragma unroll
        for (int j = 0; j < 4; ++j) rlb[lane * 4 + j] = b_[j];
      }
      store_edge_c(l2e_from(pfb), boff + p * 16);
      store_edge_c(l2e_from(rlb), boff + (l + p) * 16);
    }
  };

  // ---- sequential decode schedule (identical across batch) ----
  int dec_head = 1, i_u = 0, i_v = 0;
  const float LH = -0.69314718f;
  for (int step = 0; step < kSteps; ++step) {
    int gamma = step & 1;
    int i_t = gamma ? ++i_v : ++i_u;
    int leaf_edge = i_t + kN - 1;
    int tv = leaf_edge >> 1;
    // get_path(dec_head, tv)
    int up[12], nu = 0, down[12], nd = 0;
    {
      int c = dec_head, t = tv;
      while (c != t) {
        if (c > t) { c >>= 1; up[nu++] = c; }
        else       { down[nd++] = t; t >>= 1; }
      }
    }
    for (int i = 0; i < nu + nd; ++i) {
      int beta = (i < nu) ? up[i] : down[nd - 1 - (i - nu)];
      if (dec_head == (beta >> 1)) {
        if ((beta & 1) == 0) calc_lr(dec_head, true);
        else                 calc_lr(dec_head, false);
      } else {
        calc_parent(dec_head);
      }
      dec_head = beta;
    }
    int eoff = edge_off(leaf_edge);
    // temp = edge[leaf]
    {
      int d = lane & 15, hi = lane >> 4;
      *(v8h*)(tmpv + d * 16 + hi * 8) = *(const v8h*)(tb + (size_t)eoff * 16 + d * 16 + hi * 8);
    }
    if ((leaf_edge & 1) == 0) calc_lr(tv, true);
    else                      calc_lr(tv, false);
    // logits = e2l(top_down + temp)
    {
      int d = lane & 15, hi = lane >> 4;
      v8h a = *(const v8h*)(tb + (size_t)eoff * 16 + d * 16 + hi * 8);
      v8h b = *(const v8h*)(tmpv + d * 16 + hi * 8);
#pragma unroll
      for (int i = 0; i < 8; ++i)
        actA[(hi * 8 + i) * 32 + d] = (_Float16)((float)a[i] + (float)b[i]);
    }
    v8f clog = mlp3(32, WE1, BE1, WE2, BE2, WE3, BE3);
    {
      int n = lane & 15, mb = (lane >> 4) * 8;
      if (n < 4) {
#pragma unroll
        for (int r = 0; r < 8; ++r)
          out[step * kB * 4 + (b0 + mb + r) * 4 + n] = clog[r];
      }
    }
    // make_leaf_emb (teacher-forced)
    if (lane < 16) {
      float uu = ut[(b0 + lane) * kN + (i_t - 1)];
      float vv = vt[(b0 + lane) * kN + (i_t - 1)];
      float lp[4] = {-30.0f, -30.0f, -30.0f, -30.0f};
      if (gamma == 0) { int iu = (int)uu * 2; lp[iu] = LH; lp[iu + 1] = LH; }
      else            { int idx = (int)uu * 2 + (int)vv; lp[idx] = 0.0f; }
#pragma unroll
      for (int j = 0; j < 4; ++j) actA[lane * 32 + j] = (_Float16)lp[j];
    }
    v8f cle = mlp3(32, WG1, BG1, WG2, BG2, WG3, BG3);
    store_edge_c(cle, eoff);
  }
}

extern "C" void kernel_launch(void* const* d_in, const int* in_sizes, int n_in,
                              void* d_out, int out_size, void* d_ws, size_t ws_size,
                              hipStream_t stream) {
  // Input order (pytree-flattened setup_inputs):
  // 0:z 1:b 2:u_true 3:v_true 4:emb_z 5..10:calc_left(W,b)x3 11..16:calc_right
  // 17..22:emb2logits 23..28:logits2emb 29:no_info
  const int*   z   = (const int*)d_in[0];
  const float* u   = (const float*)d_in[2];
  const float* v   = (const float*)d_in[3];
  const float* emb = (const float*)d_in[4];
  const float* ni  = (const float*)d_in[29];
  float* outp = (float*)d_out;

  // ws layout: [edge trees: 256 tiles x 114688 f16 = 57.3 MB][Wt f16 64KB][bias f32]
  _Float16* wsE = (_Float16*)d_ws;
  _Float16* wsW = wsE + (size_t)kTiles * kTileF16;
  float*    wsB = (float*)(wsW + kWTotal);

  WPtrs wp;
  for (int i = 0; i < 24; ++i) wp.p[i] = (const float*)d_in[5 + i];

  prep_weights<<<16, 256, 0, stream>>>(wp, wsW, wsB);
  fill_aux<<<(kSteps * kB + 255) / 256, 256, 0, stream>>>(u, v, outp);
  polar_decoder<<<kBlocks, kWaves * 32, 0, stream>>>(z, u, v, emb, ni, wsW, wsB, wsE, outp);
}
